// MultiHeadAttention_76991583748704
// MI455X (gfx1250) — compile-verified, matrix operand order checked by reading, not run on hardware
//
#include <hip/hip_runtime.h>
#include <hip/hip_bf16.h>

// MI455X / gfx1250 fused multi-head attention forward.
// All matmuls: v_wmma_f32_16x16x32_f16 (f16 in, fp32 accumulate).
// HBM bound: mandatory 537MB fp32 attn write (~23us @ 23.3TB/s); attn is
// written exactly once. L2 traffic minimized: LDS-staged weight strips in the
// projection GEMMs, shared K/V fragments across q-subtiles in the fused
// attention kernel. 32x2048 fp32 score strip in LDS (263KB of 320KB/WGP).
// The 1/sqrt(dk) scale is folded into the Q projection so phase 1 is pure
// load+WMMA+store.

typedef __attribute__((ext_vector_type(16))) _Float16 v16h;
typedef __attribute__((ext_vector_type(8)))  float    v8f;

#define NH   16
#define SEQ  2048
#define DK   64
#define DM   1024
#define QT   32     // query rows per attention workgroup
#define SSTR 2056   // attn LDS row stride (floats), padded vs bank conflicts
#define BSTR 40     // proj LDS B-strip stride (halves): 80B rows, 16B aligned

union Frag {
  v16h     v;
  _Float16 h[16];
  uint4    u[2];
};

// A/B fragment for 16x16x32 f16 WMMA, per ISA layout:
// lane L: row/col = L%16, K-halves = {kb..kb+7, kb+16..kb+23}, kb = 8*(L/16).
// p points at element K=kb of a contiguous K-run (global or LDS).
template <typename PTR>
__device__ __forceinline__ v16h ld_frag_f16(PTR p) {
  Frag f;
  f.u[0] = *(const uint4*)(p);
  f.u[1] = *(const uint4*)(p + 16);
  return f.v;
}

template <typename PTR>
__device__ __forceinline__ v16h cvt_frag_f32(PTR p) {
  Frag f;
  float4 a = *(const float4*)(p);
  float4 b = *(const float4*)(p + 4);
  float4 c = *(const float4*)(p + 16);
  float4 d = *(const float4*)(p + 20);
  f.h[0]=(_Float16)a.x;  f.h[1]=(_Float16)a.y;  f.h[2]=(_Float16)a.z;  f.h[3]=(_Float16)a.w;
  f.h[4]=(_Float16)b.x;  f.h[5]=(_Float16)b.y;  f.h[6]=(_Float16)b.z;  f.h[7]=(_Float16)b.w;
  f.h[8]=(_Float16)c.x;  f.h[9]=(_Float16)c.y;  f.h[10]=(_Float16)c.z; f.h[11]=(_Float16)c.w;
  f.h[12]=(_Float16)d.x; f.h[13]=(_Float16)d.y; f.h[14]=(_Float16)d.z; f.h[15]=(_Float16)d.w;
  return f.v;
}

__device__ __forceinline__ v8f wmma16(v16h a, v16h b, v8f c) {
  return __builtin_amdgcn_wmma_f32_16x16x32_f16(false, a, false, b, (short)0, c,
                                                false, false);
}

// ---------------------------------------------------------------------------
// Wt[o][i] = (f16) W[i][o]   (1024x1024), LDS-tiled for coalescing.
// ---------------------------------------------------------------------------
__global__ void mha_w_transpose(const float* __restrict__ W,
                                _Float16* __restrict__ Wt) {
  __shared__ float tile[32][33];
  int bx = blockIdx.x * 32, by = blockIdx.y * 32;
  int tx = threadIdx.x, ty = threadIdx.y;           // 32 x 8
  for (int j = ty; j < 32; j += 8)
    tile[j][tx] = W[(size_t)(by + j) * DM + bx + tx];
  __syncthreads();
  for (int j = ty; j < 32; j += 8)
    Wt[(size_t)(bx + j) * DM + by + tx] = (_Float16)tile[tx][j];
}

// ---------------------------------------------------------------------------
// Projection GEMM: Y[t][o] = (sum_i X[t][i] * Wt[o][i] + bias[o]) * oscale
// WG = 8 waves computing a 64x128 tile; per k-chunk the 128x32 Wt strip is
// staged once in LDS and shared by all waves (B L2-traffic /4). Each wave
// owns a 16x64 subtile: 4 WMMA per k-chunk, K=1024 -> 32 chunks.
// MODE 0: Y -> f16 Qh/Kh [B,H,S,64]   MODE 1: Y -> f16 Vt [B,H,64,S]
// MODE 2: Y -> fp32 [B*S, 1024]
// ---------------------------------------------------------------------------
template <int XF16, int MODE>
__global__ __launch_bounds__(256)
void mha_proj(const void* __restrict__ Xv,
              const _Float16* __restrict__ Wt, const float* __restrict__ bias,
              void* __restrict__ Out, float oscale) {
  __shared__ _Float16 bs[128 * BSTR];                // 10,240 B

  const int mblk = blockIdx.x;         // 0..63   (4096 rows / 64)
  const int nblk = blockIdx.y;         // 0..7    (1024 cols / 128)
  const int tid  = threadIdx.x;
  const int lane = tid & 31;
  const int wave = __builtin_amdgcn_readfirstlane(tid >> 5);   // SGPR-uniform
  const int ln = lane & 15, hh = lane >> 4, kb = 8 * hh;
  const int wm = wave & 3, wn = wave >> 2;           // wave tile in 4x2 grid
  const int row = mblk * 64 + wm * 16 + ln;          // A row for this lane

  const v8f z = {0.f, 0.f, 0.f, 0.f, 0.f, 0.f, 0.f, 0.f};
  v8f acc[4] = {z, z, z, z};

  const float*    Xf = (const float*)Xv;
  const _Float16* Xh = (const _Float16*)Xv;

  // cooperative staging assignment: thread -> (o-row, 16-half chunk)
  const int so = tid >> 1;                           // 0..127
  const int sk = (tid & 1) * 16;                     // 0 or 16
  const _Float16* wsrc = Wt + (size_t)(nblk * 128 + so) * DM + sk;

  for (int kc = 0; kc < 32; ++kc) {
    __syncthreads();                                 // prior reads done
    {
      const _Float16* p = wsrc + kc * 32;
      uint4 lo = *(const uint4*)(p);
      uint4 hi = *(const uint4*)(p + 8);
      *(uint4*)(bs + so * BSTR + sk)     = lo;
      *(uint4*)(bs + so * BSTR + sk + 8) = hi;
    }
    __syncthreads();                                 // strip visible

    const int koff = kc * 32 + kb;
    v16h A;
    if (XF16) A = ld_frag_f16(Xh + (size_t)row * DM + koff);
    else      A = cvt_frag_f32(Xf + (size_t)row * DM + koff);
#pragma unroll
    for (int sub = 0; sub < 4; ++sub) {
      v16h B = ld_frag_f16(bs + (wn * 64 + sub * 16 + ln) * BSTR + kb);
      acc[sub] = wmma16(A, B, acc[sub]);
    }
  }

#pragma unroll
  for (int sub = 0; sub < 4; ++sub) {
    const int o = nblk * 128 + wn * 64 + sub * 16 + ln;
    const float bv = bias[o];
#pragma unroll
    for (int r = 0; r < 8; ++r) {
      const int t   = mblk * 64 + wm * 16 + r + 8 * hh;
      const float y = (acc[sub][r] + bv) * oscale;
      if (MODE == 2) {
        ((float*)Out)[(size_t)t * DM + o] = y;
      } else {
        const int b = t >> 11, s = t & (SEQ - 1);
        const int hd = o >> 6, oo = o & (DK - 1);
        if (MODE == 0)
          ((_Float16*)Out)[((size_t)(b * NH + hd) * SEQ + s) * DK + oo] = (_Float16)y;
        else
          ((_Float16*)Out)[((size_t)(b * NH + hd) * DK + oo) * SEQ + s] = (_Float16)y;
      }
    }
  }
}

// ---------------------------------------------------------------------------
// Fused attention: one workgroup (16 waves) per (b, h, 32-query tile).
// Phase 1: scores = (Q/8) K^T into LDS; each K B-fragment feeds both
//          q-subtiles (4 WMMA per load pair).
// Phase 2: row softmax; probs written once to global attn; kept in LDS.
// Phase 3: ctx = P @ V; each V B-fragment feeds both q-subtiles.
// ---------------------------------------------------------------------------
__global__ __launch_bounds__(512)
void mha_attn(const _Float16* __restrict__ Qh, const _Float16* __restrict__ Kh,
              const _Float16* __restrict__ Vt, float* __restrict__ attn,
              _Float16* __restrict__ ctx) {
  __shared__ float sc[QT * SSTR];                    // 263,168 bytes

  const int qt = blockIdx.x;                         // 0..63
  const int h  = blockIdx.y;                         // 0..15
  const int b  = blockIdx.z;                         // 0..1
  const int tid = threadIdx.x;
  const int lane = tid & 31;
  const int wave = __builtin_amdgcn_readfirstlane(tid >> 5);   // SGPR-uniform
  const int ln = lane & 15, hh = lane >> 4, kb = 8 * hh;

  const _Float16* Qb = Qh + ((size_t)(b * NH + h) * SEQ + qt * QT) * DK;
  const _Float16* Kb = Kh + (size_t)(b * NH + h) * SEQ * DK;
  const _Float16* Vb = Vt + (size_t)(b * NH + h) * DK * SEQ;

  const v8f z = {0.f, 0.f, 0.f, 0.f, 0.f, 0.f, 0.f, 0.f};

  // ---- Phase 1: scores -> LDS -------------------------------------------
  {
    // Q A-fragments: both q-subtiles x both d-chunks, loop-invariant
    const v16h A00 = ld_frag_f16(Qb + (ln)      * DK + 0  + kb);
    const v16h A01 = ld_frag_f16(Qb + (ln)      * DK + 32 + kb);
    const v16h A10 = ld_frag_f16(Qb + (16 + ln) * DK + 0  + kb);
    const v16h A11 = ld_frag_f16(Qb + (16 + ln) * DK + 32 + kb);
    for (int kt = wave; kt < SEQ / 16; kt += 16) {
      const int k0 = kt * 16;
      __builtin_prefetch(Kb + (size_t)(k0 + 256 + ln) * DK, 0, 1); // speculative
      v16h B0 = ld_frag_f16(Kb + (size_t)(k0 + ln) * DK + 0  + kb);
      v16h B1 = ld_frag_f16(Kb + (size_t)(k0 + ln) * DK + 32 + kb);
      v8f a0 = z, a1 = z;
      a0 = wmma16(A00, B0, a0);
      a1 = wmma16(A10, B0, a1);
      a0 = wmma16(A01, B1, a0);
      a1 = wmma16(A11, B1, a1);
#pragma unroll
      for (int r = 0; r < 8; ++r) {
        sc[(r + 8 * hh) * SSTR + k0 + ln]      = a0[r];
        sc[(16 + r + 8 * hh) * SSTR + k0 + ln] = a1[r];
      }
    }
  }
  __syncthreads();

  // ---- Phase 2: softmax (wave w handles rows 2w, 2w+1 of 32) -------------
  float* arow0 = attn + ((size_t)(b * NH + h) * SEQ + qt * QT) * SEQ;
#pragma unroll
  for (int rr = 0; rr < 2; ++rr) {
    const int m = wave * 2 + rr;
    float* row = sc + m * SSTR;

    float mx = -3.0e30f;
    for (int k = lane; k < SEQ; k += 32) mx = fmaxf(mx, row[k]);
#pragma unroll
    for (int off = 16; off > 0; off >>= 1) mx = fmaxf(mx, __shfl_xor(mx, off, 32));

    float sum = 0.f;
    for (int k = lane; k < SEQ; k += 32) {
      float p = __expf(row[k] - mx);
      row[k] = p;
      sum += p;
    }
#pragma unroll
    for (int off = 16; off > 0; off >>= 1) sum += __shfl_xor(sum, off, 32);
    const float inv = 1.0f / sum;

    float* gout = arow0 + (size_t)m * SEQ;
    for (int k4 = lane * 4; k4 < SEQ; k4 += 128) {
      float4 p = *(float4*)(row + k4);
      p.x *= inv; p.y *= inv; p.z *= inv; p.w *= inv;
      *(float4*)(row + k4) = p;
      *(float4*)(gout + k4) = p;       // the one-and-only attn write
    }
  }
  __syncthreads();

  // ---- Phase 3: ctx = P @ V (waves 0..3, one 16-col d-tile each;
  //               V fragment shared across both q-subtiles) ----------------
  if (wave < 4) {
    const int dcol = wave * 16 + ln;                 // output d column
    v8f a0 = z, a1 = z;
    for (int c = 0; c < SEQ / 32; ++c) {
      v16h B  = ld_frag_f16(Vb + (size_t)dcol * SEQ + c * 32 + kb);
      v16h A0 = cvt_frag_f32(sc + (ln)      * SSTR + c * 32 + kb);
      v16h A1 = cvt_frag_f32(sc + (16 + ln) * SSTR + c * 32 + kb);
      a0 = wmma16(A0, B, a0);
      a1 = wmma16(A1, B, a1);
    }
#pragma unroll
    for (int r = 0; r < 8; ++r) {
      const int m = r + 8 * hh;
      const size_t t0 = (size_t)b * SEQ + qt * QT + m;
      ctx[t0 * DM + h * DK + wave * 16 + ln]        = (_Float16)a0[r];
      ctx[(t0 + 16) * DM + h * DK + wave * 16 + ln] = (_Float16)a1[r];
    }
  }
}

// ---------------------------------------------------------------------------
extern "C" void kernel_launch(void* const* d_in, const int* in_sizes, int n_in,
                              void* d_out, int out_size, void* d_ws, size_t ws_size,
                              hipStream_t stream) {
  (void)in_sizes; (void)n_in; (void)out_size; (void)ws_size;

  const float* q   = (const float*)d_in[0];
  const float* k   = (const float*)d_in[1];
  const float* v   = (const float*)d_in[2];
  const float* W_q = (const float*)d_in[3];
  const float* b_q = (const float*)d_in[4];
  const float* W_k = (const float*)d_in[5];
  const float* b_k = (const float*)d_in[6];
  const float* W_v = (const float*)d_in[7];
  const float* b_v = (const float*)d_in[8];
  const float* W_o = (const float*)d_in[9];
  const float* b_o = (const float*)d_in[10];

  float* out  = (float*)d_out;                                   // [2,2048,1024]
  float* attn = out + (size_t)2 * SEQ * DM;                      // [2,16,2048,2048]

  const size_t MB = 1u << 20;
  char* ws = (char*)d_ws;
  _Float16* Wt_q = (_Float16*)(ws + 0 * MB);   // 2MB each
  _Float16* Wt_k = (_Float16*)(ws + 2 * MB);
  _Float16* Wt_v = (_Float16*)(ws + 4 * MB);
  _Float16* Wt_o = (_Float16*)(ws + 6 * MB);
  _Float16* Qh   = (_Float16*)(ws + 8 * MB);   // 8MB  [B,H,S,64] f16 (pre-scaled 1/8)
  _Float16* Kh   = (_Float16*)(ws + 16 * MB);  // 8MB
  _Float16* Vt   = (_Float16*)(ws + 24 * MB);  // 8MB  [B,H,64,S] f16
  _Float16* ctx  = (_Float16*)(ws + 32 * MB);  // 8MB  [B*S,1024] f16

  dim3 tb(32, 8), tg(DM / 32, DM / 32);
  mha_w_transpose<<<tg, tb, 0, stream>>>(W_q, Wt_q);
  mha_w_transpose<<<tg, tb, 0, stream>>>(W_k, Wt_k);
  mha_w_transpose<<<tg, tb, 0, stream>>>(W_v, Wt_v);
  mha_w_transpose<<<tg, tb, 0, stream>>>(W_o, Wt_o);

  dim3 pg(2 * SEQ / 64, DM / 128);   // 64 x 8 workgroups of 8 waves
  mha_proj<0, 0><<<pg, 256, 0, stream>>>(q, Wt_q, b_q, Qh, 0.125f); // 1/sqrt(dk)
  mha_proj<0, 0><<<pg, 256, 0, stream>>>(k, Wt_k, b_k, Kh, 1.0f);
  mha_proj<0, 1><<<pg, 256, 0, stream>>>(v, Wt_v, b_v, Vt, 1.0f);

  dim3 ag(SEQ / QT, NH, 2);
  mha_attn<<<ag, 512, 0, stream>>>(Qh, Kh, Vt, attn, ctx);

  mha_proj<1, 2><<<pg, 256, 0, stream>>>(ctx, Wt_o, b_o, out, 1.0f);
}
